// SRSGNN_47579647705747
// MI455X (gfx1250) — matrix operand order repaired and checked by reading, not compiled, not run on hardware
//
#include <hip/hip_runtime.h>
#include <hip/hip_bf16.h>

// ---------------------------------------------------------------------------
// SR-SGNN forward on MI455X (gfx1250, wave32, WMMA).
// Heavy GEMMs -> v_wmma_f32_16x16x32_bf16 (f32 accum). Elementwise GRU / LIF /
// LayerNorm fused into GEMM epilogues via LDS. ein/eout stored transposed so
// the graph-GEMM B-fragments are contiguous per lane.
// ---------------------------------------------------------------------------

#define B_   512
#define L_   200
#define D_   128
#define T_   4
#define LP_  208            // L padded to 13*16 (zero-filled)
#define M_   (B_ * L_)      // 102400 flattened rows

typedef __attribute__((ext_vector_type(16))) __bf16 v16bf;
typedef __attribute__((ext_vector_type(8)))  float  v8f;

struct B32x8 { uint4 lo, hi; };   // 32 bytes == one v16bf fragment

__device__ __forceinline__ unsigned short f2bf(float f) {
  unsigned u = __float_as_uint(f);
  u += 0x7fffu + ((u >> 16) & 1u);          // round-to-nearest-even
  return (unsigned short)(u >> 16);
}

__device__ __forceinline__ v16bf frag_zero16() {
  B32x8 z; z.lo = make_uint4(0,0,0,0); z.hi = make_uint4(0,0,0,0);
  return __builtin_bit_cast(v16bf, z);
}

// A-fragment (16x32 bf16, row-major src): lane holds row (lane&15),
// contiguous 16 K-values at k0+(lane>>4)*16.
__device__ __forceinline__ v16bf frag_a_bf16(const unsigned short* p, int ld,
                                             int row, int k0, int lane) {
  const unsigned short* r = p + (size_t)row * ld + k0 + ((lane >> 4) << 4);
  B32x8 f;
  f.lo = *(const uint4*)r;
  f.hi = *(const uint4*)(r + 8);
  return __builtin_bit_cast(v16bf, f);
}

// B-fragment (32x16 bf16) from [N][K] row-major: lane holds column n0+(lane&15),
// contiguous 16 K-values at k0+(lane>>4)*16.
__device__ __forceinline__ v16bf frag_b_bf16(const unsigned short* p, int ld,
                                             int n0, int k0, int lane, bool ok) {
  if (!ok) return frag_zero16();
  const unsigned short* r = p + (size_t)(n0 + (lane & 15)) * ld + k0 + ((lane >> 4) << 4);
  B32x8 f;
  f.lo = *(const uint4*)r;
  f.hi = *(const uint4*)(r + 8);
  return __builtin_bit_cast(v16bf, f);
}

// A-fragment with on-the-fly f32 -> bf16 conversion + OOB guards.
__device__ __forceinline__ v16bf frag_a_f32cvt(const float* rowptr, int k0, int lane,
                                               bool row_ok, bool c0_ok, bool c1_ok) {
  int h = (lane >> 4) << 4;
  float4 a0 = make_float4(0.f,0.f,0.f,0.f), a1 = a0, b0 = a0, b1 = a0;
  if (row_ok && c0_ok) {
    const float* p = rowptr + k0 + h;
    a0 = *(const float4*)p;  a1 = *(const float4*)(p + 4);
  }
  if (row_ok && c1_ok) {
    const float* p = rowptr + k0 + 8 + h;
    b0 = *(const float4*)p;  b1 = *(const float4*)(p + 4);
  }
  union { v16bf v; unsigned short u[16]; } r;
  r.u[0]=f2bf(a0.x); r.u[1]=f2bf(a0.y); r.u[2]=f2bf(a0.z); r.u[3]=f2bf(a0.w);
  r.u[4]=f2bf(a1.x); r.u[5]=f2bf(a1.y); r.u[6]=f2bf(a1.z); r.u[7]=f2bf(a1.w);
  r.u[8]=f2bf(b0.x); r.u[9]=f2bf(b0.y); r.u[10]=f2bf(b0.z); r.u[11]=f2bf(b0.w);
  r.u[12]=f2bf(b1.x); r.u[13]=f2bf(b1.y); r.u[14]=f2bf(b1.z); r.u[15]=f2bf(b1.w);
  return r.v;
}

__device__ __forceinline__ v8f wmma_bf16(v16bf a, v16bf b, v8f c) {
  return __builtin_amdgcn_wmma_f32_16x16x32_bf16(false, a, false, b, (short)0, c,
                                                 false, false);
}

__device__ __forceinline__ v8f v8f_zero() {
  v8f z = {0.f,0.f,0.f,0.f,0.f,0.f,0.f,0.f};
  return z;
}

// ---------------------------------------------------------------------------
// K0: generic f32 -> bf16 cast (weights)
__global__ void k_cast_bf16(const float* __restrict__ src,
                            unsigned short* __restrict__ dst, int n) {
  int i = blockIdx.x * 256 + threadIdx.x;
  if (i < n) dst[i] = f2bf(src[i]);
}

// K1: embedding gather -> hidden (f32 + bf16)
__global__ void k_gather(const int* __restrict__ items, const float* __restrict__ emb,
                         float* __restrict__ hf, unsigned short* __restrict__ hb) {
  int m = blockIdx.x * 2 + (threadIdx.x >> 7);
  int d = threadIdx.x & 127;
  int it = items[m];
  float v = emb[(size_t)it * D_ + d];
  hf[(size_t)m * D_ + d] = v;
  hb[(size_t)m * D_ + d] = f2bf(v);
}

// K2: ein = hidden@W_inT + b_in ; eout = hidden@W_outT + b_out
// Stored TRANSPOSED as [B][128][LP_] bf16 so they work as B-operands later.
// Block: 8 waves, M-tile 64. Waves 0-3 -> W_in, waves 4-7 -> W_out.
__global__ void __launch_bounds__(256) k_ein_eout(
    const unsigned short* __restrict__ hb,
    const unsigned short* __restrict__ winb, const unsigned short* __restrict__ woutb,
    const float* __restrict__ b_in, const float* __restrict__ b_out,
    unsigned short* __restrict__ ein_t, unsigned short* __restrict__ eout_t) {
  int tid = threadIdx.x, w = tid >> 5, lane = tid & 31;
  int which = w >> 2;
  int mt = blockIdx.x * 64 + (w & 3) * 16;
  const unsigned short* W = which ? woutb : winb;
  const float* bias = which ? b_out : b_in;
  unsigned short* dst = which ? eout_t : ein_t;

  v8f acc[8];
#pragma unroll
  for (int i = 0; i < 8; ++i) acc[i] = v8f_zero();

#pragma unroll
  for (int kt = 0; kt < 4; ++kt) {
    v16bf a = frag_a_bf16(hb, D_, mt + (lane & 15), kt * 32, lane);
#pragma unroll
    for (int nt = 0; nt < 8; ++nt) {
      v16bf b = frag_b_bf16(W, D_, nt * 16, kt * 32, lane, true);
      acc[nt] = wmma_bf16(a, b, acc[nt]);
    }
  }
  int h8 = (lane >> 4) << 3;
#pragma unroll
  for (int nt = 0; nt < 8; ++nt) {
    int n = nt * 16 + (lane & 15);
    float bv = bias[n];
#pragma unroll
    for (int v = 0; v < 8; ++v) {
      int m = mt + v + h8;
      int bb = m / L_;
      int l  = m - bb * L_;
      dst[((size_t)bb * D_ + n) * LP_ + l] = f2bf(acc[nt][v] + bv);
    }
  }
}

// K3: inputs[b,m,0:128]   = A_in[b]  @ ein[b]  + b_iah
//     inputs[b,m,128:256] = A_out[b] @ eout[b] + b_ioh   (bf16)
// Grid (13 M-tiles, 512 batches); waves 0-3 -> "in" half, 4-7 -> "out" half.
__global__ void __launch_bounds__(256) k_graph_gemm(
    const float* __restrict__ A,
    const unsigned short* __restrict__ ein_t, const unsigned short* __restrict__ eout_t,
    const float* __restrict__ b_iah, const float* __restrict__ b_ioh,
    unsigned short* __restrict__ inputs) {
  int tid = threadIdx.x, w = tid >> 5, lane = tid & 31;
  int half = w >> 2, q = w & 3;
  int b = blockIdx.y, mt = blockIdx.x * 16;
  int row = mt + (lane & 15);
  bool row_ok = row < L_;
  const float* rowptr = A + (size_t)b * (L_ * 2 * L_) + (size_t)row * (2 * L_) + half * L_;
  const unsigned short* Bsrc = (half ? eout_t : ein_t) + (size_t)b * D_ * LP_;
  const float* bias = half ? b_ioh : b_iah;

  v8f acc[2] = { v8f_zero(), v8f_zero() };

  for (int kt = 0; kt < 7; ++kt) {
    int k0 = kt * 32;
    if (row_ok && kt < 6) __builtin_prefetch(rowptr + k0 + 32, 0, 0);
    bool c0 = (k0 + ((lane >> 4) << 4)) < L_;
    bool c1 = (k0 + 8 + ((lane >> 4) << 4)) < L_;
    v16bf a = frag_a_f32cvt(rowptr, k0, lane, row_ok, c0, c1);
    bool bok = (k0 + ((lane >> 4) << 4)) < LP_;   // pad region [200,208) is zeroed
#pragma unroll
    for (int t = 0; t < 2; ++t) {
      v16bf bf = frag_b_bf16(Bsrc, LP_, q * 32 + t * 16, k0, lane, bok);
      acc[t] = wmma_bf16(a, bf, acc[t]);
    }
  }
  int h8 = (lane >> 4) << 3;
#pragma unroll
  for (int t = 0; t < 2; ++t) {
    int n = q * 32 + t * 16 + (lane & 15);
    float bv = bias[n];
#pragma unroll
    for (int v = 0; v < 8; ++v) {
      int m = mt + v + h8;
      if (m < L_)
        inputs[(((size_t)b * L_ + m) * 256) + half * 128 + n] = f2bf(acc[t][v] + bv);
    }
  }
}

// K4: gi = inputs@w_ihT + b_ih, gh = hidden@w_hhT + b_hh, fused GRU cell.
// Block = one 16-row M-tile; wave w owns N-tiles {3w,3w+1,3w+2} of both gi & gh.
// Results staged in LDS, then elementwise GRU updates hidden in place (f32+bf16).
__global__ void __launch_bounds__(256) k_gru(
    const unsigned short* __restrict__ inputs, const unsigned short* __restrict__ wihb,
    const unsigned short* __restrict__ whhb,
    const float* __restrict__ b_ih, const float* __restrict__ b_hh,
    float* __restrict__ hf, unsigned short* __restrict__ hb) {
  __shared__ float lgi[16][385];
  __shared__ float lgh[16][385];
  int tid = threadIdx.x, w = tid >> 5, lane = tid & 31;
  int m0 = blockIdx.x * 16;

  v8f ai[3], ah[3];
#pragma unroll
  for (int j = 0; j < 3; ++j) { ai[j] = v8f_zero(); ah[j] = v8f_zero(); }

#pragma unroll
  for (int kt = 0; kt < 8; ++kt) {
    v16bf a = frag_a_bf16(inputs, 256, m0 + (lane & 15), kt * 32, lane);
#pragma unroll
    for (int j = 0; j < 3; ++j) {
      v16bf bb = frag_b_bf16(wihb, 256, (3 * w + j) * 16, kt * 32, lane, true);
      ai[j] = wmma_bf16(a, bb, ai[j]);
    }
  }
#pragma unroll
  for (int kt = 0; kt < 4; ++kt) {
    v16bf a = frag_a_bf16(hb, D_, m0 + (lane & 15), kt * 32, lane);
#pragma unroll
    for (int j = 0; j < 3; ++j) {
      v16bf bb = frag_b_bf16(whhb, D_, (3 * w + j) * 16, kt * 32, lane, true);
      ah[j] = wmma_bf16(a, bb, ah[j]);
    }
  }
  int h8 = (lane >> 4) << 3;
#pragma unroll
  for (int j = 0; j < 3; ++j) {
    int n = (3 * w + j) * 16 + (lane & 15);
    float bi = b_ih[n], bh2 = b_hh[n];
#pragma unroll
    for (int v = 0; v < 8; ++v) {
      int r = v + h8;
      lgi[r][n] = ai[j][v] + bi;
      lgh[r][n] = ah[j][v] + bh2;
    }
  }
  __syncthreads();
#pragma unroll
  for (int i = 0; i < 8; ++i) {
    int idx = tid + i * 256;
    int r = idx >> 7, d = idx & 127;
    size_t g = (size_t)(m0 + r) * D_ + d;
    float ir = lgi[r][d], ii = lgi[r][d + 128], in_ = lgi[r][d + 256];
    float hr = lgh[r][d], hi = lgh[r][d + 128], hn = lgh[r][d + 256];
    float rr = 1.f / (1.f + __expf(-(ir + hr)));
    float zz = 1.f / (1.f + __expf(-(ii + hi)));
    float nn = tanhf(in_ + rr * hn);
    float ho = hf[g];
    float hnew = (1.f - zz) * ho + zz * nn;
    hf[g] = hnew;
    hb[g] = f2bf(hnew);
  }
}

// K5: ht[b] = hidden[b, alias[b, len-1]] ; htW1[b] = ht @ W1T
__global__ void k_ht(const int* __restrict__ alias_, const int* __restrict__ lens,
                     const float* __restrict__ hf, const float* __restrict__ W1,
                     float* __restrict__ ht, float* __restrict__ htw1) {
  __shared__ float s[128];
  int b = blockIdx.x, t = threadIdx.x;
  int ln = lens[b];
  int ar = alias_[b * L_ + (ln - 1)];
  float v = hf[((size_t)b * L_ + ar) * D_ + t];
  ht[b * D_ + t] = v;
  s[t] = v;
  __syncthreads();
  float acc = 0.f;
  const float* wr = W1 + (size_t)t * D_;
  for (int k = 0; k < D_; ++k) acc += wr[k] * s[k];
  htw1[b * D_ + t] = acc;
}

// K6: Qb = htW1 + seq_hidden@W2T (seq_hidden gathered+cast on the fly), then
// LayerNorm(g_q,beta_q), 4-step LIF (identical drive each step), alpha = s.W3,
// masked weighted sum of seq_hidden -> atomicAdd into a[T,B,D].
__global__ void __launch_bounds__(256) k_attn(
    const int* __restrict__ alias_, const int* __restrict__ lens,
    const float* __restrict__ hf, const unsigned short* __restrict__ w2b,
    const float* __restrict__ htw1, const float* __restrict__ W3,
    const float* __restrict__ g_q, const float* __restrict__ beta_q,
    float* __restrict__ aacc) {
  __shared__ float qb[16][129];
  __shared__ float smean[16], srstd[16];
  __shared__ float salpha[4][16];
  int tid = threadIdx.x, w = tid >> 5, lane = tid & 31;
  int b = blockIdx.y, mt = blockIdx.x * 16;
  int row = mt + (lane & 15);
  bool row_ok = row < L_;
  int ar = row_ok ? alias_[b * L_ + row] : 0;
  const float* rowptr = hf + ((size_t)b * L_ + ar) * D_;

  v8f acc = v8f_zero();
#pragma unroll
  for (int kt = 0; kt < 4; ++kt) {
    v16bf a = frag_a_f32cvt(rowptr, kt * 32, lane, row_ok, true, true);
    v16bf bb = frag_b_bf16(w2b, D_, w * 16, kt * 32, lane, true);
    acc = wmma_bf16(a, bb, acc);
  }
  int n = w * 16 + (lane & 15);
  float hw = htw1[b * D_ + n];
  int h8 = (lane >> 4) << 3;
#pragma unroll
  for (int v = 0; v < 8; ++v) qb[v + h8][n] = acc[v] + hw;
  __syncthreads();

  if (tid < 16) {                              // per-row LayerNorm stats
    float s = 0.f, s2 = 0.f;
    for (int d = 0; d < D_; ++d) { float x = qb[tid][d]; s += x; s2 += x * x; }
    float mu = s * (1.f / D_);
    float var = s2 * (1.f / D_) - mu * mu;
    smean[tid] = mu;
    srstd[tid] = rsqrtf(var + 1e-5f);
  } else if (tid < 80) {
    int i = tid - 16;
    salpha[i >> 4][i & 15] = 0.f;
  }
  __syncthreads();

  {                                            // LIF + alpha = spikes . W3
    int r = tid >> 4, sub = tid & 15;
    float mu = smean[r], rs = srstd[r];
    float pa[T_] = {0.f, 0.f, 0.f, 0.f};
#pragma unroll
    for (int j = 0; j < 8; ++j) {
      int d = sub * 8 + j;
      float x = (qb[r][d] - mu) * rs * g_q[d] + beta_q[d];
      float w3 = W3[d];
      float vv = 0.f;
#pragma unroll
      for (int t = 0; t < T_; ++t) {
        vv = 0.5f * (vv + x);                  // v += (x - v)/tau, tau=2
        float s = (vv >= 1.f) ? 1.f : 0.f;     // spike(v - 1)
        pa[t] += s * w3;
        vv -= s * vv;                          // hard reset
      }
    }
#pragma unroll
    for (int t = 0; t < T_; ++t) atomicAdd(&salpha[t][r], pa[t]);
  }
  __syncthreads();

  {                                            // a[t,b,:] += sum_m alpha*seq_h*mask
    int tg = tid >> 7;                         // 0 / 1 -> handles t = tg, tg+2
    int d = tid & 127;
    int ln = lens[b];
    for (int t2 = tg; t2 < T_; t2 += 2) {
      float s = 0.f;
      for (int r2 = 0; r2 < 16; ++r2) {
        int m = mt + r2;
        if (m < ln) {
          int ar2 = alias_[b * L_ + m];
          s += salpha[t2][r2] * hf[((size_t)b * L_ + ar2) * D_ + d];
        }
      }
      atomicAdd(&aacc[(((size_t)t2 * B_) + b) * D_ + d], s);
    }
  }
}

// K7: sh = [a | ht] -> 4-step LIF -> LayerNorm(g_sh,beta_sh) -> mean over T
//     -> out = mean_sh @ WtT.   Spikes are {0,1} so sumsq == sum.
__global__ void __launch_bounds__(256) k_final(
    const float* __restrict__ aacc, const float* __restrict__ ht,
    const float* __restrict__ g_sh, const float* __restrict__ beta_sh,
    const float* __restrict__ Wt, float* __restrict__ out) {
  __shared__ float red[256];
  __shared__ float msh[256];
  int b = blockIdx.x, k = threadIdx.x;

  float x[T_];
  if (k < 128) {
#pragma unroll
    for (int t = 0; t < T_; ++t) x[t] = aacc[(((size_t)t * B_) + b) * D_ + k];
  } else {
    float hv = ht[b * D_ + (k - 128)];
#pragma unroll
    for (int t = 0; t < T_; ++t) x[t] = hv;
  }
  float s[T_];
  float vv = 0.f;
#pragma unroll
  for (int t = 0; t < T_; ++t) {
    vv = 0.5f * (vv + x[t]);
    s[t] = (vv >= 1.f) ? 1.f : 0.f;
    vv -= s[t] * vv;
  }
  float g = g_sh[k], be = beta_sh[k];
  float accm = 0.f;
  for (int t = 0; t < T_; ++t) {
    red[k] = s[t];
    __syncthreads();
    for (int st = 128; st > 0; st >>= 1) {
      if (k < st) red[k] += red[k + st];
      __syncthreads();
    }
    float p = red[0] * (1.f / 256.f);          // mean; var = p - p^2 (binary)
    __syncthreads();
    float rs = rsqrtf(p - p * p + 1e-5f);
    accm += (s[t] - p) * rs * g + be;
  }
  msh[k] = accm * 0.25f;
  __syncthreads();
  if (k < 128) {
    const float* wr = Wt + (size_t)k * 256;
    float o = 0.f;
    for (int j = 0; j < 256; ++j) o += wr[j] * msh[j];
    out[b * D_ + k] = o;
  }
}

// ---------------------------------------------------------------------------
extern "C" void kernel_launch(void* const* d_in, const int* in_sizes, int n_in,
                              void* d_out, int out_size, void* d_ws, size_t ws_size,
                              hipStream_t stream) {
  const float* A        = (const float*)d_in[0];
  const int*   alias_   = (const int*)d_in[1];
  const int*   items    = (const int*)d_in[2];
  /* d_in[3] = mask: unused, derived from item_seq_len */
  const int*   lens     = (const int*)d_in[4];
  const float* item_emb = (const float*)d_in[5];
  const float* W_in     = (const float*)d_in[6];
  const float* b_in     = (const float*)d_in[7];
  const float* W_out    = (const float*)d_in[8];
  const float* b_out    = (const float*)d_in[9];
  const float* w_ih     = (const float*)d_in[10];
  const float* w_hh     = (const float*)d_in[11];
  const float* b_ih     = (const float*)d_in[12];
  const float* b_hh     = (const float*)d_in[13];
  const float* b_iah    = (const float*)d_in[14];
  const float* b_ioh    = (const float*)d_in[15];
  const float* W1       = (const float*)d_in[16];
  const float* W2       = (const float*)d_in[17];
  const float* W3       = (const float*)d_in[18];
  const float* Wt       = (const float*)d_in[19];
  const float* g_q      = (const float*)d_in[20];
  const float* beta_q   = (const float*)d_in[21];
  const float* g_sh     = (const float*)d_in[22];
  const float* beta_sh  = (const float*)d_in[23];

  char* ws = (char*)d_ws;
  size_t off = 0;
  auto alloc = [&](size_t bytes) -> void* {
    void* p = ws + off;
    off += (bytes + 255) & ~(size_t)255;
    return p;
  };
  float*          hf     = (float*)alloc((size_t)M_ * D_ * 4);
  unsigned short* hb     = (unsigned short*)alloc((size_t)M_ * D_ * 2);
  unsigned short* ein_t  = (unsigned short*)alloc((size_t)B_ * D_ * LP_ * 2);
  unsigned short* eout_t = (unsigned short*)alloc((size_t)B_ * D_ * LP_ * 2);
  unsigned short* inputs = (unsigned short*)alloc((size_t)M_ * 256 * 2);
  float*          ht     = (float*)alloc((size_t)B_ * D_ * 4);
  float*          htw1   = (float*)alloc((size_t)B_ * D_ * 4);
  float*          aacc   = (float*)alloc((size_t)T_ * B_ * D_ * 4);
  unsigned short* winb   = (unsigned short*)alloc(128 * 128 * 2);
  unsigned short* woutb  = (unsigned short*)alloc(128 * 128 * 2);
  unsigned short* wihb   = (unsigned short*)alloc(384 * 256 * 2);
  unsigned short* whhb   = (unsigned short*)alloc(384 * 128 * 2);
  unsigned short* w2b    = (unsigned short*)alloc(128 * 128 * 2);

  // zero pad regions + atomic accumulators (graph-capture safe)
  hipMemsetAsync(ein_t,  0, (size_t)B_ * D_ * LP_ * 2, stream);
  hipMemsetAsync(eout_t, 0, (size_t)B_ * D_ * LP_ * 2, stream);
  hipMemsetAsync(aacc,   0, (size_t)T_ * B_ * D_ * 4, stream);

  // weight casts
  k_cast_bf16<<<(16384 + 255) / 256, 256, 0, stream>>>(W_in,  winb,  16384);
  k_cast_bf16<<<(16384 + 255) / 256, 256, 0, stream>>>(W_out, woutb, 16384);
  k_cast_bf16<<<(98304 + 255) / 256, 256, 0, stream>>>(w_ih,  wihb,  98304);
  k_cast_bf16<<<(49152 + 255) / 256, 256, 0, stream>>>(w_hh,  whhb,  49152);
  k_cast_bf16<<<(16384 + 255) / 256, 256, 0, stream>>>(W2,    w2b,   16384);

  k_gather<<<M_ / 2, 256, 0, stream>>>(items, item_emb, hf, hb);

  k_ein_eout<<<M_ / 64, 256, 0, stream>>>(hb, winb, woutb, b_in, b_out, ein_t, eout_t);

  k_graph_gemm<<<dim3(13, B_), 256, 0, stream>>>(A, ein_t, eout_t, b_iah, b_ioh, inputs);

  k_gru<<<M_ / 16, 256, 0, stream>>>(inputs, wihb, whhb, b_ih, b_hh, hf, hb);

  k_ht<<<B_, 128, 0, stream>>>(alias_, lens, hf, W1, ht, htw1);

  k_attn<<<dim3(13, B_), 256, 0, stream>>>(alias_, lens, hf, w2b, htw1, W3,
                                           g_q, beta_q, aacc);

  k_final<<<B_, 256, 0, stream>>>(aacc, ht, g_sh, beta_sh, Wt, (float*)d_out);
}
